// DecomposeNetwork_48773648613488
// MI455X (gfx1250) — compile-verified, matrix operand order checked by reading, not run on hardware
//
#include <hip/hip_runtime.h>
#include <hip/hip_bf16.h>
#include <cstdint>

// ---------------- problem constants (match reference) ----------------
#define BB   16384
#define NM_  2000
#define NA_  2000
#define E1_  500
#define E2_  500
#define E3_  2000
#define NZ_  64
#define CC   4096
// BETA=0.25, ALPHA=1.0 -> forward loss factor (1+BETA)=1.25

static inline int padup(int x) { return (x + 127) & ~127; }   // host-side pad

// ---------------- WMMA types ----------------
typedef __attribute__((ext_vector_type(16))) __bf16 bf16x16;
typedef __attribute__((ext_vector_type(8)))  float  f32x8;

union FragU { bf16x16 v; uint4 q[2]; };

__device__ __forceinline__ unsigned short f2bf(float f) {
  union { float f; unsigned int u; } c; c.f = f;
  unsigned int u = c.u;
  u += 0x7FFFu + ((u >> 16) & 1u);   // round-to-nearest-even to bf16
  return (unsigned short)(u >> 16);
}

__device__ __forceinline__ int imin(int a, int b) { return a < b ? a : b; }

#define BM 128
#define BN 128
#define BK 32
#define LSTR 40   // LDS row stride in bf16 elems (80B, multiple of 16B)

// ---- gfx1250 async global->LDS path (guarded: falls back to sync copies) ----
#if __has_builtin(__builtin_amdgcn_global_load_async_to_lds_b128)
#define ASYNC_LDS 1
typedef int v4i __attribute__((vector_size(16)));
typedef __attribute__((address_space(1))) v4i as1_v4i;
typedef __attribute__((address_space(3))) v4i as3_v4i;
__device__ __forceinline__ void async_cp16(const unsigned short* g, unsigned short* l) {
  __builtin_amdgcn_global_load_async_to_lds_b128(
      (as1_v4i*)(void*)g, (as3_v4i*)(void*)l, 0, 0);
}
__device__ __forceinline__ void wait_async0() {
#if __has_builtin(__builtin_amdgcn_s_wait_asynccnt)
  __builtin_amdgcn_s_wait_asynccnt(0);
#else
  asm volatile("s_wait_asynccnt 0x0" ::: "memory");
#endif
}
#else
#define ASYNC_LDS 0
#endif

// D = act(A[M,K] @ W[K,N] + bias[N])
//   A : bf16, row stride Ksp (multiple of 128, zero-padded beyond K)
//   Wt: bf16, transposed weights [Npad, Ksp], zero-padded
//   out: bf16 (row stride OutStride, pad cols written as 0)  or  f32 (stride OutStride=N)
// M is a multiple of BM for all launches (M == 16384).
template <bool RELU, bool OUTBF16>
__global__ __launch_bounds__(256)
void gemm_bf16(const unsigned short* __restrict__ A,
               const unsigned short* __restrict__ Wt,
               const float* __restrict__ bias,
               void* __restrict__ outp,
               int M, int N, int K, int Ksp, int OutStride)
{
  const int tid  = threadIdx.x;
  const int lane = tid & 31;
  const int wave = tid >> 5;
  const int wm   = wave >> 2;      // 0..1  (64-row band)
  const int wn   = wave & 3;       // 0..3  (32-col band)
  const int m0   = blockIdx.y * BM;
  const int n0   = blockIdx.x * BN;

  f32x8 acc[4][2];
  #pragma unroll
  for (int i = 0; i < 4; i++)
    #pragma unroll
    for (int j = 0; j < 2; j++)
      #pragma unroll
      for (int e = 0; e < 8; e++) acc[i][j][e] = 0.0f;

  // copy loader: thread t -> tile row t>>1 (both A and Wt), 16 elems at (t&1)*16
  const int r = tid >> 1;
  const int h = (tid & 1) << 4;
  const unsigned short* Arow = A  + (size_t)(m0 + r) * Ksp + h;
  const unsigned short* Brow = Wt + (size_t)(n0 + r) * Ksp + h;

  const int hi8 = (lane >> 4) * 8;   // K sub-chunk select per WMMA 16-bit layout
  const int lr  = lane & 15;

  // fragment-load + WMMA stage from given LDS tile bases
  auto compute = [&](const unsigned short* AsP, const unsigned short* BsP) {
    FragU af[4], bfg[2];
    #pragma unroll
    for (int i = 0; i < 4; i++) {
      int row = wm * 64 + i * 16 + lr;
      af[i].q[0] = *(const uint4*)&AsP[row * LSTR + hi8];        // K {0..7} / {8..15}
      af[i].q[1] = *(const uint4*)&AsP[row * LSTR + hi8 + 16];   // K {16..23}/{24..31}
    }
    #pragma unroll
    for (int j = 0; j < 2; j++) {
      int coln = wn * 32 + j * 16 + lr;
      bfg[j].q[0] = *(const uint4*)&BsP[coln * LSTR + hi8];
      bfg[j].q[1] = *(const uint4*)&BsP[coln * LSTR + hi8 + 16];
    }
    #pragma unroll
    for (int i = 0; i < 4; i++)
      #pragma unroll
      for (int j = 0; j < 2; j++)
        acc[i][j] = __builtin_amdgcn_wmma_f32_16x16x32_bf16(
            false, af[i].v, false, bfg[j].v, (short)0, acc[i][j], false, false);
  };

#if ASYNC_LDS
  // double-buffered pipeline fed by global_load_async_to_lds_b128 (ASYNCcnt)
  __shared__ unsigned short As[2][BM * LSTR];
  __shared__ unsigned short Bs[2][BM * LSTR];

  auto issue = [&](int nb, int k0) {
    async_cp16(Arow + k0,     &As[nb][r * LSTR + h]);
    async_cp16(Arow + k0 + 8, &As[nb][r * LSTR + h + 8]);
    async_cp16(Brow + k0,     &Bs[nb][r * LSTR + h]);
    async_cp16(Brow + k0 + 8, &Bs[nb][r * LSTR + h + 8]);
  };

  const int nt = (K + BK - 1) / BK;   // tail tiles read zero-filled pad
  issue(0, 0);
  wait_async0();
  __syncthreads();
  for (int t = 0; t < nt; t++) {
    const int cur = t & 1;
    if (t + 1 < nt) issue(cur ^ 1, (t + 1) * BK);
    if (t + 2 < nt) {                 // global_prefetch_b8, two tiles ahead
      __builtin_prefetch(Arow + (t + 2) * BK, 0, 1);
      __builtin_prefetch(Brow + (t + 2) * BK, 0, 1);
    }
    compute(As[cur], Bs[cur]);
    wait_async0();
    __syncthreads();
  }
#else
  // sync fallback: reg-staged copies (proven path)
  __shared__ unsigned short As[BM * LSTR];
  __shared__ unsigned short Bs[BM * LSTR];

  uint4 sa0, sa1, sb0, sb1;
  {
    const uint4* p = (const uint4*)(Arow);
    sa0 = p[0]; sa1 = p[1];
    const uint4* q = (const uint4*)(Brow);
    sb0 = q[0]; sb1 = q[1];
  }
  for (int k0 = 0; k0 < K; k0 += BK) {
    *(uint4*)&As[r * LSTR + h]     = sa0;
    *(uint4*)&As[r * LSTR + h + 8] = sa1;
    *(uint4*)&Bs[r * LSTR + h]     = sb0;
    *(uint4*)&Bs[r * LSTR + h + 8] = sb1;
    __syncthreads();
    if (k0 + BK < K) {
      const uint4* p = (const uint4*)(Arow + k0 + BK);
      sa0 = p[0]; sa1 = p[1];
      const uint4* q = (const uint4*)(Brow + k0 + BK);
      sb0 = q[0]; sb1 = q[1];
    }
    if (k0 + 2 * BK < K) {
      __builtin_prefetch(Arow + k0 + 2 * BK, 0, 1);
      __builtin_prefetch(Brow + k0 + 2 * BK, 0, 1);
    }
    compute(As, Bs);
    __syncthreads();
  }
#endif

  // epilogue: C/D layout -> VGPR e: M = e + 8*(lane>=16), N = lane&15
  #pragma unroll
  for (int i = 0; i < 4; i++) {
    int rbase = m0 + wm * 64 + i * 16 + hi8;
    #pragma unroll
    for (int j = 0; j < 2; j++) {
      int col = n0 + wn * 32 + j * 16 + lr;
      if (OUTBF16) {
        unsigned short* Cb = (unsigned short*)outp;
        float bvs = bias[imin(col, N - 1)];
        bool ok = col < N;
        #pragma unroll
        for (int e = 0; e < 8; e++) {
          float v = acc[i][j][e] + bvs;
          if (RELU) v = fmaxf(v, 0.0f);
          v = ok ? v : 0.0f;                       // zero-fill pad columns
          Cb[(size_t)(rbase + e) * OutStride + col] = f2bf(v);
        }
      } else {
        float* Cf = (float*)outp;
        if (col < N) {
          float bvs = bias[col];
          #pragma unroll
          for (int e = 0; e < 8; e++) {
            float v = acc[i][j][e] + bvs;
            if (RELU) v = fmaxf(v, 0.0f);
            Cf[(size_t)(rbase + e) * OutStride + col] = v;
          }
        }
      }
    }
  }
}

// ---------------- one-time conversion passes ----------------
// dst[r, c] (bf16, stride Cp) = c < C ? src[r, c] : 0
__global__ void cvt_pad_bf16_kernel(const float* __restrict__ src,
                                    unsigned short* __restrict__ dst,
                                    int C, int Cp, long long total)
{
  long long idx = (long long)blockIdx.x * blockDim.x + threadIdx.x;
  if (idx >= total) return;
  int rrow = (int)(idx / Cp);
  int c    = (int)(idx % Cp);
  float v = (c < C) ? src[(size_t)rrow * C + c] : 0.0f;
  dst[idx] = f2bf(v);
}

// wt[n, k] (bf16, [Np, Kp]) = (n<N && k<K) ? w[k, n] : 0
__global__ void cvt_w_transpose_kernel(const float* __restrict__ w,
                                       unsigned short* __restrict__ wt,
                                       int K, int N, int Kp, long long total)
{
  long long idx = (long long)blockIdx.x * blockDim.x + threadIdx.x;
  if (idx >= total) return;
  int n = (int)(idx / Kp);
  int k = (int)(idx % Kp);
  float v = (n < N && k < K) ? w[(size_t)k * N + n] : 0.0f;
  wt[idx] = f2bf(v);
}

// zb[b, c] (bf16, stride 128) = c < 64 ? zp[b,c] + zs[b,c] : 0
__global__ void add_z_bf16_kernel(const float* __restrict__ zp,
                                  const float* __restrict__ zs,
                                  unsigned short* __restrict__ zb)
{
  int idx = blockIdx.x * blockDim.x + threadIdx.x;   // BB * 128
  int rrow = idx >> 7;
  int c    = idx & 127;
  float v = 0.0f;
  if (c < NZ_) {
    size_t o = (size_t)rrow * NZ_ + c;
    v = zp[o] + zs[o];
  }
  zb[idx] = f2bf(v);
}

__global__ void zero2_kernel(float* p)
{
  if (threadIdx.x < 2) p[threadIdx.x] = 0.0f;
}

// one thread per row: farthest-code argmax (faithful to reference) + MSE accum
__global__ __launch_bounds__(256)
void quantize_loss_kernel(const float* __restrict__ z, const float* __restrict__ cb,
                          float* __restrict__ acc)
{
  int row = blockIdx.x * blockDim.x + threadIdx.x;
  if (row >= BB) return;
  float zr[NZ_];
  const float* zp = z + (size_t)row * NZ_;
  #pragma unroll
  for (int j = 0; j < NZ_; j++) zr[j] = zp[j];

  // d_i = |z|^2 + |c_i|^2 - 2 z.c_i ; |z|^2 constant per row -> rank by |c|^2 - 2 z.c
  float best = -3.402823466e38f;
  int bidx = 0;
  for (int i = 0; i < CC; i++) {
    const float* c = cb + (size_t)i * NZ_;
    float dot = 0.0f, c2 = 0.0f;
    #pragma unroll 16
    for (int j = 0; j < NZ_; j++) {
      float cv = c[j];
      dot = fmaf(zr[j], cv, dot);
      c2  = fmaf(cv, cv, c2);
    }
    float d = c2 - 2.0f * dot;
    if (d > best) { best = d; bidx = i; }   // strict '>' == first-max (argmax ties)
  }
  const float* c = cb + (size_t)bidx * NZ_;
  float s = 0.0f;
  #pragma unroll
  for (int j = 0; j < NZ_; j++) {
    float diff = c[j] - zr[j];
    s = fmaf(diff, diff, s);
  }
  atomicAdd(acc, s);
}

__global__ void finalize_loss_kernel(const float* __restrict__ acc, float* __restrict__ out)
{
  if (threadIdx.x == 0)
    out[0] = 1.25f * (acc[0] + acc[1]) * (1.0f / (float)(BB * NZ_));
}

// ---------------- host-side orchestration ----------------
extern "C" void kernel_launch(void* const* d_in, const int* in_sizes, int n_in,
                              void* d_out, int out_size, void* d_ws, size_t ws_size,
                              hipStream_t stream)
{
  (void)in_sizes; (void)n_in; (void)ws_size;

  // setup_inputs() dict insertion order, params flattened in insertion order:
  const float* x_m   = (const float*)d_in[0];
  const float* x_a   = (const float*)d_in[1];
  const float* inm_w = (const float*)d_in[2];
  const float* inm_b = (const float*)d_in[3];
  const float* ina_w = (const float*)d_in[4];
  const float* ina_b = (const float*)d_in[5];
  const float* encm[6]; const float* enca[6]; const float* encs[6]; const float* decs[6];
  for (int j = 0; j < 6; j++) encm[j] = (const float*)d_in[6 + j];
  for (int j = 0; j < 6; j++) enca[j] = (const float*)d_in[12 + j];
  for (int j = 0; j < 6; j++) encs[j] = (const float*)d_in[18 + j];
  for (int j = 0; j < 6; j++) decs[j] = (const float*)d_in[24 + j];
  const float* decm_w = (const float*)d_in[30];
  const float* decm_b = (const float*)d_in[31];
  const float* deca_w = (const float*)d_in[32];
  const float* deca_b = (const float*)d_in[33];
  const float* codebook = (const float*)d_in[34];

  // padded dims
  const int P500  = padup(500);    // 512
  const int P2000 = padup(2000);   // 2048
  const int P64   = padup(64);     // 128

  // ---- workspace bump allocator (bf16 units) ----
  unsigned short* wsu = (unsigned short*)d_ws;
  size_t off = 0;
  auto alloc = [&](size_t n) { unsigned short* p = wsu + off; off += n; return p; };

  unsigned short* xb_m = alloc((size_t)BB * P2000);
  unsigned short* xb_a = alloc((size_t)BB * P2000);
  unsigned short* h_m  = alloc((size_t)BB * P500);
  unsigned short* h_a  = alloc((size_t)BB * P500);
  unsigned short* t1b  = alloc((size_t)BB * P500);    // enc t1 / dec d2
  unsigned short* t2b  = alloc((size_t)BB * P2000);   // enc t2 / dec d1
  unsigned short* d3b  = alloc((size_t)BB * P500);
  unsigned short* z_mb = alloc((size_t)BB * P64);
  unsigned short* z_ab = alloc((size_t)BB * P64);

  // transposed bf16 weights [Npad, Kpad]
  unsigned short* wt_inm = alloc((size_t)P500 * P2000);
  unsigned short* wt_ina = alloc((size_t)P500 * P2000);
  unsigned short* wt_em[3]; unsigned short* wt_ea[3]; unsigned short* wt_es[3];
  for (int e = 0; e < 3; e++) { // enc layer shapes: (500,500),(500,2000),(2000,64)
    size_t sz[3] = {(size_t)P500 * P500, (size_t)P2000 * P500, (size_t)P64 * P2000};
    wt_em[e] = alloc(sz[e]); wt_ea[e] = alloc(sz[e]); wt_es[e] = alloc(sz[e]);
  }
  unsigned short* wt_d1 = alloc((size_t)P2000 * P64);    // (64,2000)
  unsigned short* wt_d2 = alloc((size_t)P500 * P2000);   // (2000,500)
  unsigned short* wt_d3 = alloc((size_t)P500 * P500);    // (500,500)
  unsigned short* wt_hm = alloc((size_t)P2000 * P500);   // (500,2000)
  unsigned short* wt_ha = alloc((size_t)P2000 * P500);
  float* acc = (float*)(wsu + off);                      // off is even -> 4B aligned

  // outputs, flattened in return order
  float* out     = (float*)d_out;
  float* z_m_p   = out;
  float* z_m_s   = out + 1 * (size_t)BB * NZ_;
  float* z_a_p   = out + 2 * (size_t)BB * NZ_;
  float* z_a_s   = out + 3 * (size_t)BB * NZ_;
  float* x_m_hat = out + 4 * (size_t)BB * NZ_;
  float* x_a_hat = x_m_hat + (size_t)BB * NM_;
  float* loss    = out + ((size_t)out_size - 1);

  // ---- one-time conversions ----
  auto cvt_pad = [&](const float* src, unsigned short* dst, int C, int Cp, int R) {
    long long tot = (long long)R * Cp;
    cvt_pad_bf16_kernel<<<dim3((unsigned)((tot + 255) / 256)), dim3(256), 0, stream>>>(
        src, dst, C, Cp, tot);
  };
  auto cvt_w = [&](const float* w, unsigned short* wt, int K, int N) {
    int Kp = padup(K);
    long long tot = (long long)padup(N) * Kp;
    cvt_w_transpose_kernel<<<dim3((unsigned)((tot + 255) / 256)), dim3(256), 0, stream>>>(
        w, wt, K, N, Kp, tot);
  };

  cvt_pad(x_m, xb_m, NM_, P2000, BB);
  cvt_pad(x_a, xb_a, NA_, P2000, BB);
  cvt_w(inm_w, wt_inm, 2000, 500);
  cvt_w(ina_w, wt_ina, 2000, 500);
  {
    const int ek[3] = {500, 500, 2000}, en[3] = {500, 2000, 64};
    for (int e = 0; e < 3; e++) {
      cvt_w(encm[2 * e], wt_em[e], ek[e], en[e]);
      cvt_w(enca[2 * e], wt_ea[e], ek[e], en[e]);
      cvt_w(encs[2 * e], wt_es[e], ek[e], en[e]);
    }
  }
  cvt_w(decs[0], wt_d1, 64, 2000);
  cvt_w(decs[2], wt_d2, 2000, 500);
  cvt_w(decs[4], wt_d3, 500, 500);
  cvt_w(decm_w, wt_hm, 500, 2000);
  cvt_w(deca_w, wt_ha, 500, 2000);

  // ---- GEMM launcher ----
  auto gemm_b = [&](const unsigned short* A, int Ksp, const unsigned short* Wt,
                    const float* b, unsigned short* C, int N, int K, int OutStride) {
    dim3 g(padup(N) / BN, BB / BM);
    gemm_bf16<true, true><<<g, dim3(256), 0, stream>>>(A, Wt, b, (void*)C, BB, N, K, Ksp, OutStride);
  };
  auto gemm_f = [&](const unsigned short* A, int Ksp, const unsigned short* Wt,
                    const float* b, float* C, int N, int K) {
    dim3 g(padup(N) / BN, BB / BM);
    gemm_bf16<false, false><<<g, dim3(256), 0, stream>>>(A, Wt, b, (void*)C, BB, N, K, Ksp, N);
  };

  // input projections
  gemm_b(xb_m, P2000, wt_inm, inm_b, h_m, 500, 2000, P500);
  gemm_b(xb_a, P2000, wt_ina, ina_b, h_a, 500, 2000, P500);

  // encoders: Linear-ReLU, Linear-ReLU, Linear (f32 z straight into d_out)
  auto encoder = [&](const unsigned short* h, unsigned short* const* wt,
                     const float* const* wb, float* zout) {
    gemm_b(h,   P500,  wt[0], wb[1], t1b, 500, 500, P500);
    gemm_b(t1b, P500,  wt[1], wb[3], t2b, 2000, 500, P2000);
    gemm_f(t2b, P2000, wt[2], wb[5], zout, 64, 2000);
  };
  { unsigned short* w3[3] = {wt_em[0], wt_em[1], wt_em[2]}; encoder(h_m, w3, encm, z_m_p); }
  { unsigned short* w3[3] = {wt_ea[0], wt_ea[1], wt_ea[2]}; encoder(h_a, w3, enca, z_a_p); }
  { unsigned short* w3[3] = {wt_es[0], wt_es[1], wt_es[2]}; encoder(h_m, w3, encs, z_m_s); }
  { unsigned short* w3[3] = {wt_es[0], wt_es[1], wt_es[2]}; encoder(h_a, w3, encs, z_a_s); }

  // z = z_p + z_s (bf16, padded to 128 cols) feeds the decoder
  {
    int n = BB * P64;
    add_z_bf16_kernel<<<dim3(n / 256), dim3(256), 0, stream>>>(z_m_p, z_m_s, z_mb);
    add_z_bf16_kernel<<<dim3(n / 256), dim3(256), 0, stream>>>(z_a_p, z_a_s, z_ab);
  }

  // codebook loss (quantized latent itself is discarded by the reference graph)
  zero2_kernel<<<dim3(1), dim3(32), 0, stream>>>(acc);
  quantize_loss_kernel<<<dim3(BB / 256), dim3(256), 0, stream>>>(z_m_s, codebook, acc + 0);
  quantize_loss_kernel<<<dim3(BB / 256), dim3(256), 0, stream>>>(z_a_s, codebook, acc + 1);

  // shared decoder (Linear-ReLU x3) + head
  auto decoder = [&](const unsigned short* zb, const unsigned short* wth,
                     const float* hb, float* xhat) {
    gemm_b(zb,  P64,   wt_d1, decs[1], t2b, 2000, 64, P2000);
    gemm_b(t2b, P2000, wt_d2, decs[3], t1b, 500, 2000, P500);
    gemm_b(t1b, P500,  wt_d3, decs[5], d3b, 500, 500, P500);
    gemm_f(d3b, P500,  wth,   hb,      xhat, 2000, 500);
  };
  decoder(z_mb, wt_hm, decm_b, x_m_hat);
  decoder(z_ab, wt_ha, deca_b, x_a_hat);

  finalize_loss_kernel<<<dim3(1), dim3(32), 0, stream>>>(acc, loss);
}
